// Decomposition_80367428043380
// MI455X (gfx1250) — matrix operand order repaired and checked by reading, not compiled, not run on hardware
//
#include <hip/hip_runtime.h>

// ---------------- CDNA5 WMMA helpers (wave32, 16x16x32 f16 -> f32) ----------

typedef __attribute__((ext_vector_type(16))) _Float16 v16h;
typedef __attribute__((ext_vector_type(8)))  _Float16 v8h;
typedef __attribute__((ext_vector_type(8)))  float    v8f;
typedef __attribute__((ext_vector_type(4)))  float    v4f;

__device__ __forceinline__ v8f wmma_f16(v16h a, v16h b, v8f c) {
  return __builtin_amdgcn_wmma_f32_16x16x32_f16(false, a, false, b, (short)0, c,
                                                false, false);
}

// ---- A fragment (16x32 MxK) from row-major f32, NO clamps (caller
// guarantees in-range). ISA 7.12.2: lane<16 -> K {0..7,16..23}+g*8 pattern;
// each lane's data = two 8-float contiguous runs -> 4x 16B loads.
__device__ __forceinline__ v16h load_A_f32_fast(const float* __restrict__ A,
                                                int lda, int m0, int k0, int lane) {
  int m = m0 + (lane & 15);
  int g = (lane >> 4) & 1;
  const float* p = A + m * lda + k0 + g * 8;
  v4f x0 = *(const v4f*)(p);
  v4f x1 = *(const v4f*)(p + 4);
  v4f x2 = *(const v4f*)(p + 16);
  v4f x3 = *(const v4f*)(p + 20);
  v16h a;
#pragma unroll
  for (int j = 0; j < 4; ++j) {
    a[j]      = (_Float16)x0[j];
    a[4 + j]  = (_Float16)x1[j];
    a[8 + j]  = (_Float16)x2[j];
    a[12 + j] = (_Float16)x3[j];
  }
  return a;
}

// ---- B fragment (32x16 KxN) from TRANSPOSED row-major [N][K] f16, NO clamps.
// Each lane's 16 elements: K {0..7,16..23}+g*8 pattern -> 2x 16B loads.
__device__ __forceinline__ v16h load_B_f16_fast(const _Float16* __restrict__ Bt,
                                                int ldb, int n0, int k0, int lane) {
  int n = n0 + (lane & 15);
  int g = (lane >> 4) & 1;
  const _Float16* p = Bt + n * ldb + k0 + g * 8;
  v8h lo = *(const v8h*)(p);
  v8h hi = *(const v8h*)(p + 16);
  v16h b;
#pragma unroll
  for (int j = 0; j < 8; ++j) { b[j] = lo[j]; b[8 + j] = hi[j]; }
  return b;
}

// ---- Clamped variants (LDS): clamp the ROW ADDRESS (always safe), load
// unconditionally, then zero the fragment via value-select (no exec branches).
__device__ __forceinline__ v16h load_A_f32_lds(const float* A, int lda, int m0,
                                               int k0, int lane, int mmax) {
  int m = m0 + (lane & 15);
  int mc = (m < mmax) ? m : 0;
  int g = (lane >> 4) & 1;
  const float* p = A + mc * lda + k0 + g * 8;
  v4f x0 = *(const v4f*)(p);
  v4f x1 = *(const v4f*)(p + 4);
  v4f x2 = *(const v4f*)(p + 16);
  v4f x3 = *(const v4f*)(p + 20);
  v16h a;
#pragma unroll
  for (int j = 0; j < 4; ++j) {
    a[j]      = (_Float16)x0[j];
    a[4 + j]  = (_Float16)x1[j];
    a[8 + j]  = (_Float16)x2[j];
    a[12 + j] = (_Float16)x3[j];
  }
  if (m >= mmax) {
#pragma unroll
    for (int j = 0; j < 16; ++j) a[j] = (_Float16)0.0f;
  }
  return a;
}

__device__ __forceinline__ v16h load_A_f16_lds(const _Float16* A, int lda, int m0,
                                               int k0, int lane, int mmax) {
  int m = m0 + (lane & 15);
  int mc = (m < mmax) ? m : 0;
  int g = (lane >> 4) & 1;
  const _Float16* p = A + mc * lda + k0 + g * 8;
  v8h x0 = *(const v8h*)(p);
  v8h x1 = *(const v8h*)(p + 16);
  v16h a;
#pragma unroll
  for (int j = 0; j < 8; ++j) { a[j] = x0[j]; a[8 + j] = x1[j]; }
  if (m >= mmax) {
#pragma unroll
    for (int j = 0; j < 16; ++j) a[j] = (_Float16)0.0f;
  }
  return a;
}

__device__ __forceinline__ v16h load_B_f16_lds(const _Float16* Bt, int ldb, int n0,
                                               int k0, int lane, int nmax) {
  int n = n0 + (lane & 15);
  int nc = (n < nmax) ? n : 0;
  int g = (lane >> 4) & 1;
  const _Float16* p = Bt + nc * ldb + k0 + g * 8;
  v8h lo = *(const v8h*)(p);
  v8h hi = *(const v8h*)(p + 16);
  v16h b;
#pragma unroll
  for (int j = 0; j < 8; ++j) { b[j] = lo[j]; b[8 + j] = hi[j]; }
  if (n >= nmax) {
#pragma unroll
    for (int j = 0; j < 16; ++j) b[j] = (_Float16)0.0f;
  }
  return b;
}

// C/D f32 16x16 layout: VGPR r: lanes 0-15 -> M=r, lanes 16-31 -> M=r+8; N=lane&15.

// ---------------- constants ----------------
#define NBATCH 2048   // N*B = 32*64
#define DIN    512
#define DSL    256
#define DHID   512
#define NSLOT  8
#define NWRD   24
#define NROWKV (NBATCH * NWRD)  // 49152

// ---------------- weight convert+transpose: W[K][N] f32 -> Wt[N][K] f16 -----
__global__ void wconv_t(const float* __restrict__ W, _Float16* __restrict__ Wt,
                        int K, int N) {
  int idx = blockIdx.x * blockDim.x + threadIdx.x;
  if (idx >= K * N) return;
  int k = idx / N, n = idx % N;
  Wt[n * K + k] = (_Float16)W[k * N + n];
}

// ---------------- initial_slots = sentence_sum @ Wbind + bbind --------------
// 32x32 per wave (2x2 register blocking), 64x64 wave-tiles -> 4096 / 8 = 512 blocks
__global__ void __launch_bounds__(256) bind_gemm(
    const float* __restrict__ ssum, const _Float16* __restrict__ Wbind_t,
    const float* __restrict__ bbind, float* __restrict__ out) {
  int lane = threadIdx.x & 31, wave = threadIdx.x >> 5;
  int wt = blockIdx.x * 8 + wave;
  int m0 = (wt >> 6) * 32, n0 = (wt & 63) * 32;
  v8f acc[2][2] = {};
#pragma unroll
  for (int ks = 0; ks < 16; ++ks) {
    v16h a0 = load_A_f32_fast(ssum, DIN, m0,      ks * 32, lane);
    v16h a1 = load_A_f32_fast(ssum, DIN, m0 + 16, ks * 32, lane);
    v16h b0 = load_B_f16_fast(Wbind_t, DIN, n0,      ks * 32, lane);
    v16h b1 = load_B_f16_fast(Wbind_t, DIN, n0 + 16, ks * 32, lane);
    acc[0][0] = wmma_f16(a0, b0, acc[0][0]);
    acc[0][1] = wmma_f16(a0, b1, acc[0][1]);
    acc[1][0] = wmma_f16(a1, b0, acc[1][0]);
    acc[1][1] = wmma_f16(a1, b1, acc[1][1]);
  }
#pragma unroll
  for (int mi = 0; mi < 2; ++mi)
#pragma unroll
    for (int ni = 0; ni < 2; ++ni)
#pragma unroll
      for (int r = 0; r < 8; ++r) {
        int m = m0 + mi * 16 + r + 8 * (lane >> 4);
        int n = n0 + ni * 16 + (lane & 15);
        out[m * 2048 + n] = acc[mi][ni][r] + bbind[n];
      }
}

// ------- FUSED k = elu(X@Wk+bk)+1  AND  v = X@Wv+bv  (f16 outputs) ---------
// One wave computes a 32x32 tile of BOTH k and v: the A-fragments of
// `sentence` (dominant HBM stream, 100 MB) are loaded once and feed 8 WMMAs.
// 1536x8 wave-tiles of 32x32 -> 12288 / 8 waves = 1536 blocks.
__global__ void __launch_bounds__(256) kv_gemm(
    const float* __restrict__ sent,
    const _Float16* __restrict__ Wk_t, const _Float16* __restrict__ Wv_t,
    const float* __restrict__ bk, const float* __restrict__ bv,
    _Float16* __restrict__ kf, _Float16* __restrict__ vf) {
  int lane = threadIdx.x & 31, wave = threadIdx.x >> 5;
  int wt = blockIdx.x * 8 + wave;
  int m0 = (wt >> 3) * 32, n0 = (wt & 7) * 32;
  v8f acck[2][2] = {};
  v8f accv[2][2] = {};
#pragma unroll
  for (int ks = 0; ks < 16; ++ks) {
    v16h a0 = load_A_f32_fast(sent, DIN, m0,      ks * 32, lane);
    v16h a1 = load_A_f32_fast(sent, DIN, m0 + 16, ks * 32, lane);
    v16h bk0 = load_B_f16_fast(Wk_t, DIN, n0,      ks * 32, lane);
    v16h bk1 = load_B_f16_fast(Wk_t, DIN, n0 + 16, ks * 32, lane);
    v16h bv0 = load_B_f16_fast(Wv_t, DIN, n0,      ks * 32, lane);
    v16h bv1 = load_B_f16_fast(Wv_t, DIN, n0 + 16, ks * 32, lane);
    acck[0][0] = wmma_f16(a0, bk0, acck[0][0]);
    acck[0][1] = wmma_f16(a0, bk1, acck[0][1]);
    acck[1][0] = wmma_f16(a1, bk0, acck[1][0]);
    acck[1][1] = wmma_f16(a1, bk1, acck[1][1]);
    accv[0][0] = wmma_f16(a0, bv0, accv[0][0]);
    accv[0][1] = wmma_f16(a0, bv1, accv[0][1]);
    accv[1][0] = wmma_f16(a1, bv0, accv[1][0]);
    accv[1][1] = wmma_f16(a1, bv1, accv[1][1]);
  }
#pragma unroll
  for (int mi = 0; mi < 2; ++mi)
#pragma unroll
    for (int ni = 0; ni < 2; ++ni)
#pragma unroll
      for (int r = 0; r < 8; ++r) {
        int m = m0 + mi * 16 + r + 8 * (lane >> 4);
        int n = n0 + ni * 16 + (lane & 15);
        float xk = acck[mi][ni][r] + bk[n];
        xk = (xk > 0.0f) ? xk + 1.0f : __expf(xk);        // elu(x)+1
        kf[m * DSL + n] = (_Float16)xk;
        vf[m * DSL + n] = (_Float16)(accv[mi][ni][r] + bv[n]);
      }
}

// ---------------- slot-attention iterations + final projection --------------
__global__ void __launch_bounds__(256)
slot_iter(const float* __restrict__ init_g,
          const _Float16* __restrict__ kf,
          const _Float16* __restrict__ vf,
          const int* __restrict__ mask_g,
          const _Float16* __restrict__ Wq_t,  const float* __restrict__ bq,
          const float* __restrict__ g_ln,     const float* __restrict__ b_ln,
          const _Float16* __restrict__ W1_t,  const float* __restrict__ b1,
          const _Float16* __restrict__ W2_t,  const float* __restrict__ b2,
          const _Float16* __restrict__ Wproj_t, const float* __restrict__ bproj,
          float* __restrict__ out) {
  __shared__ __attribute__((aligned(16))) _Float16 s_init[NSLOT * DSL];  // 4 KB
  __shared__ __attribute__((aligned(16))) float    s_slots[NSLOT * DSL]; // 8 KB
  __shared__ __attribute__((aligned(16))) _Float16 s_kA[NWRD * DSL];     // 12 KB
  __shared__ __attribute__((aligned(16))) _Float16 s_vT[DSL * 32];       // 16 KB (pad 24->32)
  __shared__ __attribute__((aligned(16))) float    s_qupd[NSLOT * DSL];  // 8 KB (q f16 / upd f32)
  __shared__ __attribute__((aligned(16))) float    s_logits[NWRD * NSLOT];
  __shared__ __attribute__((aligned(16))) _Float16 s_attnT[NSLOT * 32];
  __shared__ __attribute__((aligned(16))) _Float16 s_hA[NSLOT * DSL];    // 4 KB
  __shared__ __attribute__((aligned(16))) _Float16 s_mid[NSLOT * DHID];  // 8 KB
  __shared__ float s_maskf[NWRD];
  __shared__ float s_mean[NSLOT], s_rstd[NSLOT], s_colsum[NSLOT];

  const int t = threadIdx.x, lane = t & 31, wave = t >> 5;
  const int batch = blockIdx.x;
  _Float16* s_q = (_Float16*)s_qupd;

  // ---- load per-batch tiles into LDS ----
  for (int e = t; e < NSLOT * DSL; e += 256) {
    float x = init_g[batch * 2048 + e];
    s_init[e] = (_Float16)x;
    s_slots[e] = x;                     // slots = initial_slots
  }
  for (int e = t; e < NWRD * DSL; e += 256)
    s_kA[e] = kf[batch * NWRD * DSL + e];
  for (int e = t; e < DSL * 32; e += 256) {
    int s = e >> 5, w = e & 31;
    s_vT[e] = (w < NWRD) ? vf[batch * NWRD * DSL + w * DSL + s] : (_Float16)0.0f;
  }
  if (t < NWRD)
    s_maskf[t] = -1e6f * (1.0f - (float)mask_g[batch * NWRD + t]);
  __syncthreads();

  for (int it = 0; it < 3; ++it) {
    // ---- (a) q = elu((slots@Wq + bq + init) / 16) + 1 -> f16 [8][256] ----
#pragma unroll
    for (int j = 0; j < 2; ++j) {
      int n0 = (wave + 8 * j) * 16;
      v8f acc = {};
#pragma unroll
      for (int ks = 0; ks < 8; ++ks) {
        v16h a = load_A_f32_lds(s_slots, DSL, 0, ks * 32, lane, NSLOT);
        v16h b = load_B_f16_fast(Wq_t, DSL, n0, ks * 32, lane);
        acc = wmma_f16(a, b, acc);
      }
#pragma unroll
      for (int r = 0; r < 8; ++r) {
        int m = r + 8 * (lane >> 4), n = n0 + (lane & 15);
        if (m < NSLOT) {
          float x = (acc[r] + bq[n] + (float)s_init[m * DSL + n]) * 0.0625f;
          x = (x > 0.0f) ? x + 1.0f : __expf(x);
          s_q[m * DSL + n] = (_Float16)x;
        }
      }
    }
    __syncthreads();

    // ---- (b) logits[w][s] = sum_d k[w,d] q[s,d] + maskf[w] ----
    if (wave < 2) {
      int m0 = wave * 16;
      v8f acc = {};
#pragma unroll
      for (int ks = 0; ks < 8; ++ks) {
        v16h a = load_A_f16_lds(s_kA, DSL, m0, ks * 32, lane, NWRD);
        v16h b = load_B_f16_lds(s_q, DSL, 0, ks * 32, lane, NSLOT);
        acc = wmma_f16(a, b, acc);
      }
#pragma unroll
      for (int r = 0; r < 8; ++r) {
        int m = m0 + r + 8 * (lane >> 4), n = lane & 15;
        if (m < NWRD && n < NSLOT)
          s_logits[m * NSLOT + n] = acc[r] + s_maskf[m];
      }
    }
    __syncthreads();

    // ---- (c) softmax over slots, +eps, renormalize over words ----
    if (t < NWRD) {
      float mx = -3.4e38f;
      for (int j = 0; j < NSLOT; ++j) mx = fmaxf(mx, s_logits[t * NSLOT + j]);
      float sum = 0.0f, e[NSLOT];
      for (int j = 0; j < NSLOT; ++j) { e[j] = __expf(s_logits[t * NSLOT + j] - mx); sum += e[j]; }
      for (int j = 0; j < NSLOT; ++j) s_logits[t * NSLOT + j] = e[j] / sum + 1e-6f;
    }
    __syncthreads();
    if (t < NSLOT) {
      float cs = 0.0f;
      for (int i = 0; i < NWRD; ++i) cs += s_logits[i * NSLOT + t];
      s_colsum[t] = cs;
    }
    __syncthreads();
    {  // attnT [slot][word], K padded 24->32 with zeros
      int j = t >> 5, i = t & 31;
      s_attnT[t] = (i < NWRD) ? (_Float16)(s_logits[i * NSLOT + j] / s_colsum[j])
                              : (_Float16)0.0f;
    }
    __syncthreads();

    // ---- (d) updates = attnT @ v  (f32 into s_qupd) ----
#pragma unroll
    for (int j = 0; j < 2; ++j) {
      int n0 = (wave + 8 * j) * 16;
      v8f acc = {};
      v16h a = load_A_f16_lds(s_attnT, 32, 0, 0, lane, NSLOT);
      v16h b = load_B_f16_fast(s_vT, 32, n0, 0, lane);
      acc = wmma_f16(a, b, acc);
#pragma unroll
      for (int r = 0; r < 8; ++r) {
        int m = r + 8 * (lane >> 4), n = n0 + (lane & 15);
        if (m < NSLOT) s_qupd[m * DSL + n] = acc[r];
      }
    }
    __syncthreads();

    // ---- (e) layernorm -> s_hA f16 ----
    if (t < NSLOT) {
      float s = 0.0f, ss = 0.0f;
      for (int n = 0; n < DSL; ++n) { float x = s_qupd[t * DSL + n]; s += x; ss += x * x; }
      float mean = s * (1.0f / DSL);
      float var = ss * (1.0f / DSL) - mean * mean;
      s_mean[t] = mean;
      s_rstd[t] = rsqrtf(var + 1e-5f);
    }
    __syncthreads();
    for (int e = t; e < NSLOT * DSL; e += 256) {
      int m = e >> 8, n = e & 255;
      float x = (s_qupd[e] - s_mean[m]) * s_rstd[m] * g_ln[n] + b_ln[n];
      s_hA[e] = (_Float16)x;
    }
    __syncthreads();

    // ---- (f) mid = relu(hA @ W1 + b1) -> f16 [8][512] ----
#pragma unroll
    for (int j = 0; j < 4; ++j) {
      int n0 = (wave + 8 * j) * 16;
      v8f acc = {};
#pragma unroll
      for (int ks = 0; ks < 8; ++ks) {
        v16h a = load_A_f16_lds(s_hA, DSL, 0, ks * 32, lane, NSLOT);
        v16h b = load_B_f16_fast(W1_t, DSL, n0, ks * 32, lane);
        acc = wmma_f16(a, b, acc);
      }
#pragma unroll
      for (int r = 0; r < 8; ++r) {
        int m = r + 8 * (lane >> 4), n = n0 + (lane & 15);
        if (m < NSLOT) s_mid[m * DHID + n] = (_Float16)fmaxf(acc[r] + b1[n], 0.0f);
      }
    }
    __syncthreads();

    // ---- (g) slots += (mid @ W2 + b2) / 256 ----
#pragma unroll
    for (int j = 0; j < 2; ++j) {
      int n0 = (wave + 8 * j) * 16;
      v8f acc = {};
#pragma unroll
      for (int ks = 0; ks < 16; ++ks) {
        v16h a = load_A_f16_lds(s_mid, DHID, 0, ks * 32, lane, NSLOT);
        v16h b = load_B_f16_fast(W2_t, DHID, n0, ks * 32, lane);
        acc = wmma_f16(a, b, acc);
      }
#pragma unroll
      for (int r = 0; r < 8; ++r) {
        int m = r + 8 * (lane >> 4), n = n0 + (lane & 15);
        if (m < NSLOT)
          s_slots[m * DSL + n] += (acc[r] + b2[n]) * (1.0f / 256.0f);
      }
    }
    __syncthreads();
  }

  // ---- final: out = concat(init, slots) @ Wproj + bproj ----
  for (int e = t; e < NSLOT * DHID; e += 256) {
    int m = e >> 9, k2 = e & 511;
    s_mid[e] = (k2 < DSL) ? s_init[m * DSL + k2]
                          : (_Float16)s_slots[m * DSL + (k2 - DSL)];
  }
  __syncthreads();
#pragma unroll
  for (int j = 0; j < 2; ++j) {
    int n0 = (wave + 8 * j) * 16;
    v8f acc = {};
#pragma unroll
    for (int ks = 0; ks < 16; ++ks) {
      v16h a = load_A_f16_lds(s_mid, DHID, 0, ks * 32, lane, NSLOT);
      v16h b = load_B_f16_fast(Wproj_t, DHID, n0, ks * 32, lane);
      acc = wmma_f16(a, b, acc);
    }
#pragma unroll
    for (int r = 0; r < 8; ++r) {
      int m = r + 8 * (lane >> 4), n = n0 + (lane & 15);
      if (m < NSLOT)
        out[batch * 2048 + m * DSL + n] = acc[r] + bproj[n];
    }
  }
}

// ---------------- host launcher ----------------
extern "C" void kernel_launch(void* const* d_in, const int* in_sizes, int n_in,
                              void* d_out, int out_size, void* d_ws, size_t ws_size,
                              hipStream_t stream) {
  const float* sent  = (const float*)d_in[0];
  const float* ssum  = (const float*)d_in[1];
  const int*   mask  = (const int*)d_in[2];
  const float* Wk    = (const float*)d_in[3];
  const float* bk    = (const float*)d_in[4];
  const float* Wv    = (const float*)d_in[5];
  const float* bv    = (const float*)d_in[6];
  const float* Wq    = (const float*)d_in[7];
  const float* bq    = (const float*)d_in[8];
  const float* g_ln  = (const float*)d_in[9];
  const float* b_ln  = (const float*)d_in[10];
  const float* W1    = (const float*)d_in[11];
  const float* b1    = (const float*)d_in[12];
  const float* W2    = (const float*)d_in[13];
  const float* b2    = (const float*)d_in[14];
  const float* Wbind = (const float*)d_in[15];
  const float* bbind = (const float*)d_in[16];
  const float* Wproj = (const float*)d_in[17];
  const float* bproj = (const float*)d_in[18];
  float* out = (float*)d_out;

  char* ws = (char*)d_ws;
  size_t off = 0;
  float*    init_ws = (float*)(ws + off);    off += (size_t)NBATCH * 2048 * 4;   // 16 MB
  _Float16* kf      = (_Float16*)(ws + off); off += (size_t)NROWKV * DSL * 2;    // 24 MB
  _Float16* vf      = (_Float16*)(ws + off); off += (size_t)NROWKV * DSL * 2;    // 24 MB
  _Float16* Wk_t    = (_Float16*)(ws + off); off += (size_t)DIN * DSL * 2;
  _Float16* Wv_t    = (_Float16*)(ws + off); off += (size_t)DIN * DSL * 2;
  _Float16* Wq_t    = (_Float16*)(ws + off); off += (size_t)DSL * DSL * 2;
  _Float16* W1_t    = (_Float16*)(ws + off); off += (size_t)DSL * DHID * 2;
  _Float16* W2_t    = (_Float16*)(ws + off); off += (size_t)DHID * DSL * 2;
  _Float16* Wbind_t = (_Float16*)(ws + off); off += (size_t)DIN * 2048 * 2;
  _Float16* Wproj_t = (_Float16*)(ws + off); off += (size_t)(2 * DSL) * DSL * 2;

  // weight convert + transpose to [N][K] f16
  wconv_t<<<(DIN * DSL + 255) / 256, 256, 0, stream>>>(Wk, Wk_t, DIN, DSL);
  wconv_t<<<(DIN * DSL + 255) / 256, 256, 0, stream>>>(Wv, Wv_t, DIN, DSL);
  wconv_t<<<(DSL * DSL + 255) / 256, 256, 0, stream>>>(Wq, Wq_t, DSL, DSL);
  wconv_t<<<(DSL * DHID + 255) / 256, 256, 0, stream>>>(W1, W1_t, DSL, DHID);
  wconv_t<<<(DHID * DSL + 255) / 256, 256, 0, stream>>>(W2, W2_t, DHID, DSL);
  wconv_t<<<(DIN * 2048 + 255) / 256, 256, 0, stream>>>(Wbind, Wbind_t, DIN, 2048);
  wconv_t<<<(2 * DSL * DSL + 255) / 256, 256, 0, stream>>>(Wproj, Wproj_t, 2 * DSL, DSL);

  // initial_slots = sentence_sum @ Wbind + bbind : 64x64 wave-tiles of 32x32
  bind_gemm<<<512, 256, 0, stream>>>(ssum, Wbind_t, bbind, init_ws);

  // fused k/v projections: 1536x8 wave-tiles of 32x32, 8 waves per block
  kv_gemm<<<1536, 256, 0, stream>>>(sent, Wk_t, Wv_t, bk, bv, kf, vf);

  // slot attention iterations + projection, one workgroup per (n,b)
  slot_iter<<<NBATCH, 256, 0, stream>>>(init_ws, kf, vf, mask,
                                        Wq_t, bq, g_ln, b_ln,
                                        W1_t, b1, W2_t, b2,
                                        Wproj_t, bproj, out);
}
